// GraphSAGE_83846351552681
// MI455X (gfx1250) — compile-verified, matrix operand order checked by reading, not run on hardware
//
#include <hip/hip_runtime.h>
#include <hip/hip_bf16.h>

// ---------------------------------------------------------------------------
// GraphSAGE on MI455X (gfx1250, wave32).
//  - Edge aggregation: HBM/atomic bound -> float4 gather + global f32 atomics
//    (dest array is 25.6MB -> resolves in 192MB L2).
//  - SAGE linear step: fused [agg|x] (Mx256) @ [Wl;Wr] (256x128) GEMM using
//    v_wmma_f32_16x16x32_f16; A tile staged into LDS via async global->LDS
//    (ASYNCcnt path); B pre-packed into the ISA fragment layout
//    (CDNA5 ISA 16-bit A/B layouts) and streamed from L2.
// ---------------------------------------------------------------------------

typedef __attribute__((ext_vector_type(16))) _Float16 v16h;
typedef __attribute__((ext_vector_type(8)))  _Float16 v8h;
typedef __attribute__((ext_vector_type(8)))  float    v8f;
typedef __attribute__((ext_vector_type(4)))  int      v4i;

// Builtin param types (per hipcc diagnostic): global int4*, LDS int4*.
typedef __attribute__((address_space(1))) v4i* gv4i_p;
typedef __attribute__((address_space(3))) v4i* lv4i_p;

#define N_NODES  50000
#define N_EDGES  600000
#define FEAT     128
#define KTOT     256          // concat [agg | x]
#define MTILE    64
#define M_PAD    50048        // 782 * 64
#define KPAD     264          // LDS row pitch in halves (528B: bank-shifted)
#define N_GRAPHS 512

#if defined(__has_builtin)
#if __has_builtin(__builtin_amdgcn_global_load_async_to_lds_b128)
#define HAVE_ASYNC_LDS 1
#endif
#if __has_builtin(__builtin_amdgcn_s_wait_asynccnt)
#define HAVE_WAIT_ASYNC_BUILTIN 1
#endif
#endif

__device__ __forceinline__ void wait_asynccnt0() {
#if defined(HAVE_WAIT_ASYNC_BUILTIN)
  __builtin_amdgcn_s_wait_asynccnt(0);
#else
  asm volatile("s_wait_asynccnt 0x0" ::: "memory");
#endif
}

// ---------------------------------------------------------------- utilities
__global__ __launch_bounds__(256) void k_zero(float* __restrict__ p, long n) {
  long i = (long)blockIdx.x * blockDim.x + threadIdx.x;
  if (i < n) p[i] = 0.0f;
}

// ------------------------------------------------------------------ degrees
__global__ __launch_bounds__(256) void k_degree(const int* __restrict__ dst,
                                                float* __restrict__ deg) {
  int e = blockIdx.x * blockDim.x + threadIdx.x;
  if (e >= N_EDGES) return;
  unsigned d = (unsigned)dst[e];
  if (d < N_NODES) atomicAdd(&deg[d], 1.0f);
}

// ---------------------------------------------------- edge scatter (mean num)
// one thread = (edge, 4 features); float4 gather, 4 f32 atomics (hit in L2)
__global__ __launch_bounds__(256) void k_scatter(const int* __restrict__ src,
                                                 const int* __restrict__ dst,
                                                 const float* __restrict__ h,
                                                 float* __restrict__ agg) {
  long t = (long)blockIdx.x * blockDim.x + threadIdx.x;
  long e = t >> 5;
  if (e >= N_EDGES) return;
  int f4 = (int)(t & 31) * 4;
  unsigned s = (unsigned)src[e];
  unsigned d = (unsigned)dst[e];
  if (s >= N_NODES || d >= N_NODES) return;
  const float4 v = *(const float4*)(h + (size_t)s * FEAT + f4);
  float* a = agg + (size_t)d * FEAT + f4;
  atomicAdd(a + 0, v.x);
  atomicAdd(a + 1, v.y);
  atomicAdd(a + 2, v.z);
  atomicAdd(a + 3, v.w);
}

// ------------------------------------------- pack A = f16([agg/deg | h_in])
__global__ __launch_bounds__(256) void k_pack_a(const float* __restrict__ agg,
                                                const float* __restrict__ deg,
                                                const float* __restrict__ h,
                                                _Float16* __restrict__ A) {
  long t = (long)blockIdx.x * blockDim.x + threadIdx.x;
  if (t >= (long)M_PAD * FEAT) return;
  int m = (int)(t >> 7);
  int f = (int)(t & 127);
  float av = 0.0f, hv = 0.0f;
  if (m < N_NODES) {
    float dg = deg[m];
    av = agg[(size_t)m * FEAT + f] / fmaxf(dg, 1.0f);
    hv = h[(size_t)m * FEAT + f];
  }
  A[(size_t)m * KTOT + f]        = (_Float16)av;
  A[(size_t)m * KTOT + 128 + f]  = (_Float16)hv;
}

// -------------------------- pack B = f16([Wl;Wr]) into WMMA fragment layout
// Fragment order: [ks(8)][nt(8)][lane(32)][half(16)]
// 16-bit B 32x16 (KxN): lane<16 -> N=n0+lane, K = kbase+0..7 (halves 0..7)
// and kbase+16..23 (halves 8..15) with kbase = k0 (+8 for lanes 16..31).
__global__ __launch_bounds__(256) void k_pack_b(const float* __restrict__ Wl,
                                                const float* __restrict__ Wr,
                                                _Float16* __restrict__ B) {
  int idx = blockIdx.x * blockDim.x + threadIdx.x;   // 0..32767
  int j    = idx & 15;
  int lane = (idx >> 4) & 31;
  int nt   = (idx >> 9) & 7;
  int ks   = (idx >> 12) & 7;
  int n     = nt * 16 + (lane & 15);
  int kbase = ks * 32 + ((lane >> 4) ? 8 : 0);
  int kk    = kbase + (j < 8 ? j : j + 8);           // j-8+16 == j+8
  float w = (kk < 128) ? Wl[kk * FEAT + n] : Wr[(kk - 128) * FEAT + n];
  B[idx] = (_Float16)w;
}

// --------------------------------------------------------------- WMMA GEMM
// out[m][n] = relu( A[m][:] @ Bcat[:][n] + bias[n] ), M tiled by 64/block,
// N=128 in-block. 8 waves: wave -> (mt = w&3, nt group = (w>>2)*4 .. +3).
__global__ __launch_bounds__(256) void k_sage_gemm(
    const _Float16* __restrict__ A,      // [M_PAD][256] row-major f16
    const _Float16* __restrict__ Bfrag,  // packed fragments (8*8*32*16)
    const float* __restrict__ bias,      // [128]
    float* __restrict__ out,             // [N_NODES][128] f32
    int m_nodes) {
  __shared__ _Float16 sA[MTILE * KPAD];  // 64 rows, pitch 264 halves

  const int tid  = threadIdx.x;
  const int m0   = blockIdx.x * MTILE;
  const int lane = tid & 31;

  // Warm the packed-weight lines (global_prefetch_b8) while A is staged.
  __builtin_prefetch(Bfrag + ((size_t)tid << 7), 0, 0);

  // Stage A tile: 64 x 256 halves = 2048 16B chunks, 8 per thread.
#if defined(HAVE_ASYNC_LDS)
  {
    const _Float16* g = A + (size_t)m0 * KTOT;
#pragma unroll
    for (int i = 0; i < 8; ++i) {
      int c   = tid + i * 256;          // chunk id
      int row = c >> 5;                 // 32 chunks per row
      int c8  = c & 31;                 // 8 halves per chunk
      __builtin_amdgcn_global_load_async_to_lds_b128(
          (gv4i_p)(g + ((size_t)row << 8) + c8 * 8),
          (lv4i_p)(sA + row * KPAD + c8 * 8),
          /*offset=*/0, /*cpol=*/0);
    }
    wait_asynccnt0();                   // ASYNCcnt==0: LDS writes landed
  }
#else
  {
    const uint4* gsrc = (const uint4*)(A + (size_t)m0 * KTOT);
#pragma unroll
    for (int i = 0; i < 8; ++i) {
      int c   = tid + i * 256;
      int row = c >> 5;
      int c8  = c & 31;
      *(uint4*)(sA + row * KPAD + c8 * 8) = gsrc[(row << 5) + c8];
    }
  }
#endif
  __syncthreads();

  const int wave   = tid >> 5;
  const int mt     = wave & 3;            // 0..3
  const int ntBase = (wave >> 2) << 2;    // 0 or 4

  v8f acc[4] = {};

  // A fragment addressing (16-bit A 16x32): lane<16 holds row M=lane,
  // K = k0+0..7 and k0+16..23; lanes 16..31 hold same rows, K offset +8.
  const int rowL  = mt * 16 + (lane & 15);
  const int kHalf = (lane >> 4) * 8;

#pragma unroll
  for (int ks = 0; ks < 8; ++ks) {
    const int k0 = ks * 32;
    union { v16h v; v8h h[2]; } af;
    af.h[0] = *(const v8h*)(sA + rowL * KPAD + k0 + kHalf);
    af.h[1] = *(const v8h*)(sA + rowL * KPAD + k0 + kHalf + 16);
#pragma unroll
    for (int j = 0; j < 4; ++j) {
      const int nt = ntBase + j;
      const v16h bf = *(const v16h*)(Bfrag + (((ks * 8 + nt) * 32 + lane) << 4));
      acc[j] = __builtin_amdgcn_wmma_f32_16x16x32_f16(
          /*neg_a=*/false, af.v, /*neg_b=*/false, bf,
          /*c_mod=*/(short)0, acc[j], /*reuse_a=*/false, /*reuse_b=*/false);
    }
  }

  // Epilogue: C/D layout -> lane gives N = n0 + (lane&15); VGPR r gives
  // M = r (lanes 0..15) / r+8 (lanes 16..31). Bias + ReLU, skip pad rows.
#pragma unroll
  for (int j = 0; j < 4; ++j) {
    const int n  = (ntBase + j) * 16 + (lane & 15);
    const float bv = bias[n];
    const int mBase = m0 + mt * 16 + ((lane >> 4) * 8);
#pragma unroll
    for (int r = 0; r < 8; ++r) {
      const int m = mBase + r;
      if (m < m_nodes) {
        float v = acc[j][r] + bv;
        out[(size_t)m * FEAT + n] = v > 0.0f ? v : 0.0f;
      }
    }
  }
}

// ------------------------------------------------------------- mean pooling
__global__ __launch_bounds__(256) void k_pool(const float* __restrict__ h,
                                              const int* __restrict__ batch,
                                              float* __restrict__ psum,
                                              float* __restrict__ cnt) {
  long t = (long)blockIdx.x * blockDim.x + threadIdx.x;
  long m = t >> 5;
  if (m >= N_NODES) return;
  int f4 = (int)(t & 31) * 4;
  unsigned g = (unsigned)batch[m];
  if (g >= N_GRAPHS) return;
  const float4 v = *(const float4*)(h + (size_t)m * FEAT + f4);
  float* p = psum + (size_t)g * FEAT + f4;
  atomicAdd(p + 0, v.x);
  atomicAdd(p + 1, v.y);
  atomicAdd(p + 2, v.z);
  atomicAdd(p + 3, v.w);
  if (f4 == 0) atomicAdd(&cnt[g], 1.0f);
}

// ------------------------------------------------------------- classifier
__global__ __launch_bounds__(128) void k_fc1(const float* __restrict__ psum,
                                             const float* __restrict__ cnt,
                                             const float* __restrict__ Wf1,
                                             const float* __restrict__ bf1,
                                             float* __restrict__ emb) {
  int g = blockIdx.x, j = threadIdx.x;
  float inv = 1.0f / fmaxf(cnt[g], 1.0f);
  float acc = bf1[j];
  for (int f = 0; f < FEAT; ++f)
    acc += psum[g * FEAT + f] * inv * Wf1[f * 128 + j];
  emb[g * 128 + j] = acc;
}

__global__ __launch_bounds__(32) void k_fc2(const float* __restrict__ emb,
                                            const float* __restrict__ Wf2,
                                            const float* __restrict__ bf2,
                                            float* __restrict__ out) {
  int g = blockIdx.x, c = threadIdx.x;
  if (c >= 2) return;
  float acc = bf2[c];
  for (int j = 0; j < 128; ++j) acc += emb[g * 128 + j] * Wf2[j * 2 + c];
  out[g * 2 + c] = acc;
}

// ---------------------------------------------------------------------------
extern "C" void kernel_launch(void* const* d_in, const int* in_sizes, int n_in,
                              void* d_out, int out_size, void* d_ws, size_t ws_size,
                              hipStream_t stream) {
  const float* x     = (const float*)d_in[0];
  const int*   ei    = (const int*)d_in[1];
  const int*   srcI  = ei;
  const int*   dstI  = ei + N_EDGES;
  const int*   batch = (const int*)d_in[2];
  const float* Wl[4] = {(const float*)d_in[4], (const float*)d_in[7],
                        (const float*)d_in[10], (const float*)d_in[13]};
  const float* Wr[4] = {(const float*)d_in[5], (const float*)d_in[8],
                        (const float*)d_in[11], (const float*)d_in[14]};
  const float* bb[4] = {(const float*)d_in[6], (const float*)d_in[9],
                        (const float*)d_in[12], (const float*)d_in[15]};
  const float* Wf1 = (const float*)d_in[16];
  const float* bf1 = (const float*)d_in[17];
  const float* Wf2 = (const float*)d_in[18];
  const float* bf2 = (const float*)d_in[19];

  // ---- workspace carve (all offsets 256B aligned)
  char* ws = (char*)d_ws;
  const size_t szNode = (size_t)N_NODES * FEAT * sizeof(float);   // 25.6 MB
  float*    P     = (float*)ws;            ws += szNode;
  float*    Q     = (float*)ws;            ws += szNode;
  _Float16* Afp16 = (_Float16*)ws;         ws += (size_t)M_PAD * KTOT * 2;
  float*    deg   = (float*)ws;            ws += 200192;          // 50000 f32, padded
  _Float16* Bfrag = (_Float16*)ws;         ws += 8 * 8 * 32 * 16 * 2;
  float*    psum  = (float*)ws;            ws += (size_t)N_GRAPHS * FEAT * 4;
  float*    cnt   = (float*)ws;            ws += 2048;
  float*    emb   = (float*)ws;            ws += (size_t)N_GRAPHS * 128 * 4;

  auto zero = [&](float* p, long n) {
    k_zero<<<dim3((unsigned)((n + 255) / 256)), dim3(256), 0, stream>>>(p, n);
  };

  // ---- degrees (same graph every layer)
  zero(deg, N_NODES);
  k_degree<<<dim3((N_EDGES + 255) / 256), dim3(256), 0, stream>>>(dstI, deg);

  // ---- 4 SAGE layers: buffer plan  h_in: x,Q,P,Q   agg: P,P,Q,P   out: Q,P,Q,P
  const float* h_in[4] = {x, Q, P, Q};
  float*       aggL[4] = {P, P, Q, P};
  float*       outL[4] = {Q, P, Q, P};

  const long packN   = (long)M_PAD * FEAT;
  const unsigned gScat = (unsigned)(((long)N_EDGES * 32 + 255) / 256);
  const unsigned gPack = (unsigned)((packN + 255) / 256);
  const unsigned gGemm = M_PAD / MTILE;   // 782

  for (int l = 0; l < 4; ++l) {
    zero(aggL[l], (long)N_NODES * FEAT);
    k_scatter<<<dim3(gScat), dim3(256), 0, stream>>>(srcI, dstI, h_in[l], aggL[l]);
    k_pack_a<<<dim3(gPack), dim3(256), 0, stream>>>(aggL[l], deg, h_in[l], Afp16);
    k_pack_b<<<dim3(128), dim3(256), 0, stream>>>(Wl[l], Wr[l], Bfrag);
    k_sage_gemm<<<dim3(gGemm), dim3(256), 0, stream>>>(Afp16, Bfrag, bb[l],
                                                       outL[l], N_NODES);
  }

  // ---- global mean pool + MLP head
  zero(psum, (long)N_GRAPHS * FEAT);
  zero(cnt, N_GRAPHS);
  k_pool<<<dim3((unsigned)(((long)N_NODES * 32 + 255) / 256)), dim3(256), 0, stream>>>(
      outL[3], batch, psum, cnt);
  k_fc1<<<dim3(N_GRAPHS), dim3(128), 0, stream>>>(psum, cnt, Wf1, bf1, emb);
  k_fc2<<<dim3(N_GRAPHS), dim3(32), 0, stream>>>(emb, Wf2, bf2, (float*)d_out);
}
